// PairInteraction_72885595013261
// MI455X (gfx1250) — matrix operand order repaired and checked by reading, not compiled, not run on hardware
//
#include <hip/hip_runtime.h>
#include <hip/hip_bf16.h>

// PairInteraction for MI455X (gfx1250, wave32, WMMA).
// HBM-bound: dominant traffic is the [N,32,64] padded neighbor buffer.
// bf16 intermediates; einsum+bilinear+up-projection fused per 16-atom block;
// neighbor-slot zeroing replaced by a 1.6MB validity map (vs 205MB memset);
// streamed-once tensors use non-temporal cache hints.

#define EMB   256   // atom embedding
#define PIN    64   // pair_in
#define POUT   64   // pair_out
#define RBF    16   // rbf channels
#define KMAXN  32   // neighbor slots

typedef __attribute__((ext_vector_type(16))) __bf16        v16bf;
typedef __attribute__((ext_vector_type(8)))  float         v8f;
typedef __attribute__((ext_vector_type(4)))  unsigned int  u32x4;

union FragU {
  u32x4          u[2];
  unsigned short s[16];
  v16bf          v;
};

// f32 pair -> packed bf16x2 (truncate) in ONE v_perm_b32
static __device__ __forceinline__ unsigned int packbf(float lo, float hi) {
  return __builtin_amdgcn_perm(__float_as_uint(hi), __float_as_uint(lo),
                               0x07060302u);
}

// f32 -> bf16 (truncate); stores fold into *_store_d16_hi_b16
static __device__ __forceinline__ unsigned short bftrunc(float f) {
  return (unsigned short)(__float_as_uint(f) >> 16);
}

// round-to-nearest-even (cold path: weight prepack only)
static __device__ __forceinline__ unsigned short f2bf(float f) {
  unsigned int u = __float_as_uint(f);
  return (unsigned short)((u + 0x7FFFu + ((u >> 16) & 1u)) >> 16);
}

static __device__ __forceinline__ float silu_f(float x) {
  return x * __builtin_amdgcn_rcpf(1.0f + __expf(-x));
}

static __device__ __forceinline__ v16bf frag2(u32x4 a, u32x4 b) {
  FragU f; f.u[0] = a; f.u[1] = b; return f.v;
}

// 16 f32 -> A fragment via 8 v_perm_b32
static __device__ __forceinline__ v16bf packA(float4 a, float4 b, float4 c, float4 d) {
  u32x4 lo = {packbf(a.x, a.y), packbf(a.z, a.w), packbf(b.x, b.y), packbf(b.z, b.w)};
  u32x4 hi = {packbf(c.x, c.y), packbf(c.z, c.w), packbf(d.x, d.y), packbf(d.z, d.w)};
  return frag2(lo, hi);
}

static __device__ __forceinline__ v8f wmma_bf16(v16bf a, v16bf b, v8f c) {
  return __builtin_amdgcn_wmma_f32_16x16x32_bf16(false, a, false, b, (short)0, c,
                                                 false, false);
}

// ---------------------------------------------------------------------------
// Prepack f32 weight [K,N] (row-major, optionally *scale) into bf16 WMMA
// B-fragment layout: frag[ks][t][lane][j] = W[ks*32+8*(lane>>4)+(j&7)+16*(j>>3)][t*16+(lane&15)]
// ---------------------------------------------------------------------------
__global__ __launch_bounds__(256)
void k_prepack(const float* __restrict__ W, unsigned short* __restrict__ F,
               int K, int N, const float* __restrict__ scalep, int useScale) {
  int total = (K / 32) * (N / 16) * 512;
  int idx = blockIdx.x * blockDim.x + threadIdx.x;
  if (idx >= total) return;
  float s = useScale ? *scalep : 1.0f;
  int j    = idx & 15;
  int lane = (idx >> 4) & 31;
  int t    = (idx >> 9) % (N / 16);
  int ks   = idx / (512 * (N / 16));
  int k = ks * 32 + 8 * (lane >> 4) + (j & 7) + 16 * (j >> 3);
  int n = t * 16 + (lane & 15);
  F[idx] = f2bf(W[(size_t)k * N + n] * s);
}

// ---------------------------------------------------------------------------
// Down projection: x_b = bf16(silu(h @ W_down)), 16 atoms per 128-thread block.
// ---------------------------------------------------------------------------
__global__ __launch_bounds__(128)
void k_down(const float* __restrict__ h, const unsigned short* __restrict__ WdF,
            unsigned short* __restrict__ xb, int nAtoms) {
  __shared__ __align__(16) float lh[16 * EMB];  // 16KB atom tile
  const int tid  = threadIdx.x;
  const int wave = tid >> 5;        // n-tile owner (N=64 -> 4 tiles)
  const int lane = tid & 31;
  const int half = lane >> 4;
  const int l15  = lane & 15;
  const int atomBase = blockIdx.x * 16;

  // cooperative coalesced load of 16 rows of h into LDS (h is read once -> NT)
  for (int i = tid; i < 16 * (EMB / 4); i += 128) {
    int row = i >> 6, c4 = i & 63;
    int ga = atomBase + row; if (ga >= nAtoms) ga = nAtoms - 1;
    *(u32x4*)&lh[row * EMB + c4 * 4] = __builtin_nontemporal_load(
        (const u32x4*)(h + (size_t)ga * EMB + (size_t)c4 * 4));
  }
  __syncthreads();

  v8f acc = {};
#pragma unroll
  for (int ks = 0; ks < EMB / 32; ++ks) {
    int c0 = ks * 32 + 8 * half;
    const float* rowp = &lh[l15 * EMB];
    float4 f0 = *(const float4*)(rowp + c0);
    float4 f1 = *(const float4*)(rowp + c0 + 4);
    float4 f2 = *(const float4*)(rowp + c0 + 16);
    float4 f3 = *(const float4*)(rowp + c0 + 20);
    v16bf A = packA(f0, f1, f2, f3);
    const u32x4* bp =
        (const u32x4*)(WdF + ((((size_t)ks * 4 + wave) * 32 + lane) * 16));
    v16bf B = frag2(bp[0], bp[1]);
    acc = wmma_bf16(A, B, acc);
  }

  unsigned short* op = xb + (size_t)(atomBase + 8 * half) * PIN + wave * 16 + l15;
  if (atomBase + 16 <= nAtoms) {          // uniform fast path: no per-row guards
#pragma unroll
    for (int r = 0; r < 8; ++r)
      op[(size_t)r * PIN] = bftrunc(silu_f(acc[r]));
  } else {
    for (int r = 0; r < 8; ++r)
      if (atomBase + r + 8 * half < nAtoms)
        op[(size_t)r * PIN] = bftrunc(silu_f(acc[r]));
  }
}

// ---------------------------------------------------------------------------
// Scatter: one wave per edge; 32 lanes move 64 bf16 (128B) coalesced.
// Lane 0 also marks the slot valid (replaces the 205MB zero-fill of x2).
// Duplicate (dst,k) slots: arbitrary winner, matching jnp .at[].set semantics.
// ---------------------------------------------------------------------------
__global__ __launch_bounds__(256)
void k_scatter(const int* __restrict__ eidx, const int* __restrict__ tnk,
               const unsigned short* __restrict__ xb,
               unsigned short* __restrict__ x2,
               unsigned char* __restrict__ valid, int nEdges) {
  int gw   = (blockIdx.x * blockDim.x + threadIdx.x) >> 5;
  int lane = threadIdx.x & 31;
  int nW   = (gridDim.x * blockDim.x) >> 5;
  for (int e = gw; e < nEdges; e += nW) {
    int src = __builtin_nontemporal_load(eidx + e);            // edge_index[0][e]
    int dst = __builtin_nontemporal_load(eidx + nEdges + e);   // edge_index[1][e]
    int k   = __builtin_nontemporal_load(tnk + e);
    unsigned int v = *(const unsigned int*)(xb + (size_t)src * PIN + lane * 2);
    *(unsigned int*)(x2 + ((size_t)dst * KMAXN + k) * PIN + lane * 2) = v;
    if (lane == 0) valid[(size_t)dst * KMAXN + k] = 1;
  }
}

// ---------------------------------------------------------------------------
// Fused: per 16-atom block
//   phase 1: x_ba2[a] = rad[a](16x32) @ x2[a](32x64)        -> LDS (bf16)
//   phase 2: h_out    = x_ba2(16x1024) @ (W_bil*scale)      -> LDS (bf16)
//   phase 3: out      = silu(h_out(16x64) @ W_up(64x256))   -> HBM (f32)
// Invalid neighbor slots are zero-masked during LDS staging.
// ---------------------------------------------------------------------------
__global__ __launch_bounds__(128)
void k_fused(const float* __restrict__ rad, const unsigned short* __restrict__ x2,
             const unsigned char* __restrict__ valid,
             const unsigned short* __restrict__ WbF,
             const unsigned short* __restrict__ WuF,
             float* __restrict__ out, int nAtoms) {
  __shared__ __align__(16) unsigned short lx2[KMAXN * PIN];    // 4KB  x2 tile
  __shared__ __align__(16) float          lrad[RBF * KMAXN];   // 2KB  rad tile
  __shared__ __align__(16) unsigned short lxa[16 * RBF * PIN]; // 32KB x_ba2
  __shared__ __align__(16) unsigned short lho[16 * POUT];      // 2KB  h_out
  const int tid  = threadIdx.x;
  const int wave = tid >> 5;
  const int lane = tid & 31;
  const int half = lane >> 4;
  const int l15  = lane & 15;
  const int atomBase = blockIdx.x * 16;
  const u32x4 z4 = {0u, 0u, 0u, 0u};

  // ---- phase 1: einsum per local atom ----
  for (int a = 0; a < 16; ++a) {
    int ga = atomBase + a; if (ga >= nAtoms) ga = nAtoms - 1;
    // stage x2 tile, zero-masking invalid k rows (16B covers an eighth of a k-row)
    const u32x4* xs = (const u32x4*)(x2 + (size_t)ga * (KMAXN * PIN));
    unsigned char v0 = valid[(size_t)ga * KMAXN + (tid >> 3)];
    unsigned char v1 = valid[(size_t)ga * KMAXN + ((tid + 128) >> 3)];
    u32x4 d0 = __builtin_nontemporal_load(xs + tid);
    u32x4 d1 = __builtin_nontemporal_load(xs + tid + 128);
    ((u32x4*)lx2)[tid]       = v0 ? d0 : z4;
    ((u32x4*)lx2)[tid + 128] = v1 ? d1 : z4;
    ((u32x4*)lrad)[tid] = __builtin_nontemporal_load(
        (const u32x4*)(rad + (size_t)ga * (RBF * KMAXN)) + tid);

    // prefetch next atom's tiles while this one is consumed
    int gn = atomBase + a + 1; if (gn >= nAtoms) gn = nAtoms - 1;
    __builtin_prefetch((const u32x4*)(x2 + (size_t)gn * (KMAXN * PIN)) + tid, 0, 1);
    __builtin_prefetch((const u32x4*)(rad + (size_t)gn * (RBF * KMAXN)) + tid, 0, 1);
    __syncthreads();

    // A = rad (16 rbf x 32 k) from LDS, perm-packed to bf16
    const float* rp = &lrad[l15 * KMAXN];
    int c0 = 8 * half;
    float4 f0 = *(const float4*)(rp + c0);
    float4 f1 = *(const float4*)(rp + c0 + 4);
    float4 f2 = *(const float4*)(rp + c0 + 16);
    float4 f3 = *(const float4*)(rp + c0 + 20);
    v16bf A = packA(f0, f1, f2, f3);

    // B = x2 tile column slice (k = 8*half + (j&7) + 16*(j>>3), d = wave*16 + l15)
    FragU bf;
#pragma unroll
    for (int j = 0; j < 16; ++j)
      bf.s[j] = lx2[(8 * half + (j & 7) + 16 * (j >> 3)) * PIN + wave * 16 + l15];

    v8f acc = {};
    acc = wmma_bf16(A, bf.v, acc);
#pragma unroll
    for (int i = 0; i < 8; ++i)
      lxa[a * (RBF * PIN) + (i + 8 * half) * PIN + wave * 16 + l15] =
          bftrunc(acc[i]);
    __syncthreads();
  }

  // ---- phase 2: bilinear, M=16 atoms, K=1024, N=64 (wave owns one n-tile) ----
  v8f acc2 = {};
#pragma unroll 4
  for (int ks = 0; ks < 32; ++ks) {
    const unsigned short* ap = &lxa[l15 * (RBF * PIN) + ks * 32 + 8 * half];
    v16bf A = frag2(*(const u32x4*)ap, *(const u32x4*)(ap + 16));
    const u32x4* bp =
        (const u32x4*)(WbF + ((((size_t)ks * 4 + wave) * 32 + lane) * 16));
    v16bf B = frag2(bp[0], bp[1]);
    acc2 = wmma_bf16(A, B, acc2);
  }
#pragma unroll
  for (int i = 0; i < 8; ++i)       // scale already folded into WbF
    lho[(i + 8 * half) * POUT + wave * 16 + l15] = bftrunc(acc2[i]);
  __syncthreads();

  // ---- phase 3: up projection, M=16 atoms, K=64, N=256 (wave owns 4 n-tiles) ----
  v8f accu[4] = {{}, {}, {}, {}};
#pragma unroll
  for (int ks = 0; ks < 2; ++ks) {
    const unsigned short* ap = &lho[l15 * POUT + ks * 32 + 8 * half];
    v16bf A = frag2(*(const u32x4*)ap, *(const u32x4*)(ap + 16));
#pragma unroll
    for (int t = 0; t < 4; ++t) {
      int nt = wave * 4 + t;
      const u32x4* bp =
          (const u32x4*)(WuF + ((((size_t)ks * 16 + nt) * 32 + lane) * 16));
      v16bf B = frag2(bp[0], bp[1]);
      accu[t] = wmma_bf16(A, B, accu[t]);
    }
  }

  float* ob = out + (size_t)(atomBase + 8 * half) * EMB + l15;
  if (atomBase + 16 <= nAtoms) {    // uniform fast path; out is write-once -> NT
#pragma unroll
    for (int t = 0; t < 4; ++t)
#pragma unroll
      for (int i = 0; i < 8; ++i)
        __builtin_nontemporal_store(silu_f(accu[t][i]),
                                    ob + (size_t)i * EMB + (wave * 4 + t) * 16);
  } else {
    for (int t = 0; t < 4; ++t)
      for (int i = 0; i < 8; ++i)
        if (atomBase + i + 8 * half < nAtoms)
          ob[(size_t)i * EMB + (wave * 4 + t) * 16] = silu_f(accu[t][i]);
  }
}

// ---------------------------------------------------------------------------
static inline size_t align256(size_t x) { return (x + 255) & ~(size_t)255; }

extern "C" void kernel_launch(void* const* d_in, const int* in_sizes, int n_in,
                              void* d_out, int out_size, void* d_ws, size_t ws_size,
                              hipStream_t stream) {
  const float* h      = (const float*)d_in[0];
  const float* rad    = (const float*)d_in[1];
  const int*   eidx   = (const int*)d_in[2];
  const int*   tnk    = (const int*)d_in[3];
  const float* Wdown  = (const float*)d_in[4];
  const float* Wbil   = (const float*)d_in[5];
  const float* Wup    = (const float*)d_in[6];
  const float* scale  = (const float*)d_in[7];
  float*       out    = (float*)d_out;

  const int nAtoms = in_sizes[0] / EMB;
  const int nEdges = in_sizes[3];

  size_t off = 0;
  size_t offWd = off; off = align256(off + (size_t)(EMB / 32) * (PIN / 16) * 512 * 2);
  size_t offWb = off; off = align256(off + (size_t)((RBF * PIN) / 32) * (POUT / 16) * 512 * 2);
  size_t offWu = off; off = align256(off + (size_t)(POUT / 32) * (EMB / 16) * 512 * 2);
  size_t offXb = off; off = align256(off + (size_t)nAtoms * PIN * 2);
  size_t offVl = off; size_t vlBytes = (size_t)nAtoms * KMAXN;
  off = align256(off + vlBytes);
  size_t offX2 = off; size_t x2Bytes = (size_t)nAtoms * KMAXN * PIN * 2;
  off = align256(off + x2Bytes);
  if (off > ws_size) return;

  char* ws = (char*)d_ws;
  unsigned short* WdF = (unsigned short*)(ws + offWd);
  unsigned short* WbF = (unsigned short*)(ws + offWb);
  unsigned short* WuF = (unsigned short*)(ws + offWu);
  unsigned short* xb  = (unsigned short*)(ws + offXb);
  unsigned char*  vl  = (unsigned char*)(ws + offVl);
  unsigned short* x2  = (unsigned short*)(ws + offX2);

  // 1) weight prepack (scale folded into the bilinear weights)
  k_prepack<<<(EMB / 32) * (PIN / 16) * 2, 256, 0, stream>>>(Wdown, WdF, EMB, PIN, scale, 0);
  k_prepack<<<((RBF * PIN) / 32) * (POUT / 16) * 2, 256, 0, stream>>>(Wbil, WbF, RBF * PIN, POUT, scale, 1);
  k_prepack<<<(POUT / 32) * (EMB / 16) * 2, 256, 0, stream>>>(Wup, WuF, POUT, EMB, scale, 0);

  // 2) clear only the validity map (1.6MB) instead of zero-filling x2 (205MB)
  hipMemsetAsync(vl, 0, vlBytes, stream);

  // 3) down projection
  k_down<<<(nAtoms + 15) / 16, 128, 0, stream>>>(h, WdF, xb, nAtoms);

  // 4) edge scatter (after memset; stream order gives the dependency)
  k_scatter<<<4096, 256, 0, stream>>>(eidx, tnk, xb, x2, vl, nEdges);

  // 5) fused einsum + bilinear + up-projection
  k_fused<<<(nAtoms + 15) / 16, 128, 0, stream>>>(rad, x2, vl, WbF, WuF, out, nAtoms);
}
